// MultiHeadAttention_20727512171045
// MI455X (gfx1250) — matrix operand order repaired and checked by reading, not compiled, not run on hardware
//
#include <hip/hip_runtime.h>
#include <hip/hip_bf16.h>
#include <stdint.h>

#define D_MODEL 1024
#define HEADS   16
#define D_HEAD  64
#define SEQ     2048
#define BATCH   4
#define MTOT    (BATCH*SEQ)   // 8192 rows

typedef __bf16 bf16;
typedef __attribute__((ext_vector_type(16))) __bf16        v16bf;
typedef __attribute__((ext_vector_type(8)))  float         v8f;
typedef __attribute__((ext_vector_type(4)))  unsigned int  u32x4;

union FragU {
    u32x4        q[2];
    unsigned int w[8];
    bf16         e[16];
    v16bf        v;
};

// ---------------------------------------------------------------------------
// gfx1250 async global->LDS copy path (ASYNCcnt), with safe fallback.
// Builtin pointees are 4 x int GCC vectors, src AS(1), dst AS(3).
// ---------------------------------------------------------------------------
#if defined(__AMDGCN__) && __has_builtin(__builtin_amdgcn_global_load_async_to_lds_b128)
#define USE_ASYNC 1
typedef int v4i_gcc __attribute__((vector_size(16)));
typedef __attribute__((address_space(1))) v4i_gcc* async_gptr_t;
typedef __attribute__((address_space(3))) v4i_gcc* async_lptr_t;
#else
#define USE_ASYNC 0
#endif

static __device__ __forceinline__ void cp16_g2l(bf16* l, const bf16* g) {
#if USE_ASYNC
    __builtin_amdgcn_global_load_async_to_lds_b128(
        (async_gptr_t)(uintptr_t)g,
        (async_lptr_t)(uintptr_t)l,
        /*offset=*/0, /*cpol=*/0);
#else
    *(u32x4*)l = *(const u32x4*)g;
#endif
}

static __device__ __forceinline__ void async_fence() {
#if USE_ASYNC
#if __has_builtin(__builtin_amdgcn_s_wait_asynccnt)
    __builtin_amdgcn_s_wait_asynccnt(0);
#else
    asm volatile("s_wait_asynccnt 0" ::: "memory");
#endif
#endif
}

// ---------------------------------------------------------------------------
// gfx1250 Tensor Data Mover (TENSORcnt) for strided 2D bf16 tiles -> LDS,
// with row padding done by the DMA engine (32 dwords data + 4 dwords pad
// per row => 72-element LDS row stride). This toolchain exposes the 6-arg
// builtin: (u32x4 g0, i32x8 g1, i32x4 g2, i32x4 g3, i32x8 g4, i32 cpol).
// ---------------------------------------------------------------------------
#if defined(__AMDGCN__) && __has_builtin(__builtin_amdgcn_tensor_load_to_lds) && \
    __has_builtin(__builtin_amdgcn_s_wait_tensorcnt)
#define USE_TDM 1
typedef int      ti32x4 __attribute__((vector_size(16)));
typedef int      ti32x8 __attribute__((vector_size(32)));
typedef unsigned tu32x4 __attribute__((vector_size(16)));

static __device__ __forceinline__ void tdm_load_2d(unsigned lds_off, const bf16* g,
                                                   int tile_w, int tile_h,
                                                   int row_stride /*elems*/) {
    uint64_t ga = (uint64_t)(uintptr_t)g;
    tu32x4 g0 = {
        1u,                                                 // D# count=1, user mode
        lds_off,                                            // lds_addr (bytes)
        (unsigned)(ga & 0xFFFFFFFFu),                       // global_addr[31:0]
        (unsigned)((ga >> 32) & 0x01FFFFFFu) | (2u << 30)   // global_addr[56:32] | type=2
    };
    ti32x8 g1 = {
        // data_size=1 (2B) | pad_enable | pad_interval=4 (32 dw) | pad_amount=3 (4 dw)
        (int)((1u << 16) | (1u << 20) | (4u << 22) | (3u << 25)),
        (int)(((unsigned)row_stride & 0xFFFFu) << 16),      // tensor_dim0[15:0]
        (int)((((unsigned)row_stride >> 16) & 0xFFFFu) |
              (16384u << 16)),                              // dim0[31:16] | tensor_dim1[15:0]
        (int)(((unsigned)tile_w & 0xFFFFu) << 16),          // tensor_dim1[31:16]=0 | tile_dim0
        (int)((unsigned)tile_h & 0xFFFFu),                  // tile_dim1 | tile_dim2=0
        (int)(unsigned)row_stride,                          // tensor_dim0_stride[31:0]
        0,                                                  // stride[47:32] | dim1_stride lo
        0
    };
    ti32x4 g2 = {0, 0, 0, 0};
    ti32x4 g3 = {0, 0, 0, 0};
    ti32x8 g4 = {0, 0, 0, 0, 0, 0, 0, 0};
    __builtin_amdgcn_tensor_load_to_lds(g0, g1, g2, g3, g4, /*cpol=*/0);
}

static __device__ __forceinline__ void tdm_wait() {
    __builtin_amdgcn_s_wait_tensorcnt(0);
}
#else
#define USE_TDM 0
#endif

static __device__ __forceinline__ bf16 to_bf16(float f) {
    unsigned u = __float_as_uint(f);
    u += 0x7FFFu + ((u >> 16) & 1u);           // round-to-nearest-even
    unsigned short h = (unsigned short)(u >> 16);
    bf16 r;
    __builtin_memcpy(&r, &h, 2);
    return r;
}

static __device__ __forceinline__ v8f vzero8() {
    v8f z;
#pragma unroll
    for (int i = 0; i < 8; ++i) z[i] = 0.0f;
    return z;
}

static __device__ __forceinline__ v8f wmma_bf16(v16bf a, v16bf b, v8f c) {
    return __builtin_amdgcn_wmma_f32_16x16x32_bf16(false, a, false, b,
                                                   (short)0, c, false, false);
}

// ---------------------------------------------------------------------------
// fp32 -> bf16 conversion (optionally scaled; 1/sqrt(D_HEAD) folded into Wq)
// ---------------------------------------------------------------------------
__global__ __launch_bounds__(256)
void cvt_f32_bf16(const float* __restrict__ x, bf16* __restrict__ y,
                  int n, float scale) {
    int i = (blockIdx.x * 256 + threadIdx.x) * 4;
    if (i >= n) return;
    float4 v = *(const float4*)(x + i);
    unsigned short h0, h1, h2, h3;
    bf16 b0 = to_bf16(v.x * scale); __builtin_memcpy(&h0, &b0, 2);
    bf16 b1 = to_bf16(v.y * scale); __builtin_memcpy(&h1, &b1, 2);
    bf16 b2 = to_bf16(v.z * scale); __builtin_memcpy(&h2, &b2, 2);
    bf16 b3 = to_bf16(v.w * scale); __builtin_memcpy(&h3, &b3, 2);
    uint64_t packed = (uint64_t)h0 | ((uint64_t)h1 << 16) |
                      ((uint64_t)h2 << 32) | ((uint64_t)h3 << 48);
    *(uint64_t*)(y + i) = packed;
}

// ---------------------------------------------------------------------------
// Y[M,N] = X[M,K] * W[N,K]^T   (torch Linear, no bias), bf16 in, bf16/f32 out
// Tile: 128x128x32, 256 threads = 8 waves (2x4), wave tile 64x32 = 8 WMMA acc
// Global->LDS via async b128 copies (double buffered), fragments via ds_load.
// ---------------------------------------------------------------------------
template <int WRITE_F32>
__global__ __launch_bounds__(256)
void gemm_xwt(const bf16* __restrict__ X, const bf16* __restrict__ W,
              bf16* __restrict__ Yb, float* __restrict__ Yf) {
    constexpr int BM = 128, BN = 128, BK = 32, LD = 40; // LD pad: bank-safe, 16B aligned rows
    __shared__ bf16 As[2][BM * LD];
    __shared__ bf16 Bs[2][BN * LD];

    const int tid  = threadIdx.x;
    const int lane = tid & 31;
    const int wave = tid >> 5;
    const int ln   = lane & 15;   // N / M index inside 16
    const int lh   = lane >> 4;   // lane half
    const int kb8  = lh * 8;      // A K base per half is 2*kb8 = 16*lh
    const int wm   = wave & 1;    // 2 waves along M
    const int wn   = wave >> 1;   // 4 waves along N
    const int Mb   = blockIdx.x * BM;
    const int Nb   = blockIdx.y * BN;

    v8f acc[4][2];
#pragma unroll
    for (int i = 0; i < 4; ++i)
#pragma unroll
        for (int j = 0; j < 2; ++j) acc[i][j] = vzero8();

    auto load_tile = [&](int kt, int buf) {
        const int kpos = kt * BK;
#pragma unroll
        for (int i = 0; i < 2; ++i) {
            int linear = i * 256 + tid;          // 512 chunks of 8 elems
            int row = linear >> 2;
            int c8  = (linear & 3) * 8;
            cp16_g2l(&As[buf][row * LD + c8],
                     &X[(size_t)(Mb + row) * D_MODEL + kpos + c8]);
            cp16_g2l(&Bs[buf][row * LD + c8],
                     &W[(size_t)(Nb + row) * D_MODEL + kpos + c8]);
        }
    };

    load_tile(0, 0);
    async_fence();
    __syncthreads();

    for (int kt = 0; kt < D_MODEL / BK; ++kt) {
        const int cur = kt & 1;
        if (kt + 1 < D_MODEL / BK) load_tile(kt + 1, cur ^ 1);
        if (kt + 2 < D_MODEL / BK) { // gfx1250 global_prefetch path
            int row = tid >> 2, c8 = (tid & 3) * 8;
            __builtin_prefetch(&X[(size_t)(Mb + row) * D_MODEL + (kt + 2) * BK + c8], 0, 1);
            __builtin_prefetch(&W[(size_t)(Nb + row) * D_MODEL + (kt + 2) * BK + c8], 0, 1);
        }

        FragU af[4];
#pragma unroll
        for (int sm = 0; sm < 4; ++sm) {           // A frag: M=row, K=16*lh+e
            const bf16* base = &As[cur][(wm * 64 + sm * 16 + ln) * LD];
#pragma unroll
            for (int p = 0; p < 8; ++p) {
                int K = (kb8 << 1) + (p << 1);
                af[sm].w[p] = *(const unsigned int*)&base[K];
            }
        }
        FragU bfrag[2];
#pragma unroll
        for (int sn = 0; sn < 2; ++sn) {           // B frag: N=ln, K=16*lh+e
            const bf16* base = &Bs[cur][(wn * 32 + sn * 16 + ln) * LD + lh * 16];
            bfrag[sn].q[0] = *(const u32x4*)&base[0];
            bfrag[sn].q[1] = *(const u32x4*)&base[8];
        }
#pragma unroll
        for (int sm = 0; sm < 4; ++sm)
#pragma unroll
            for (int sn = 0; sn < 2; ++sn)
                acc[sm][sn] = wmma_bf16(af[sm].v, bfrag[sn].v, acc[sm][sn]);

        async_fence();     // next tile's async copies have landed in LDS
        __syncthreads();
    }

#pragma unroll
    for (int sm = 0; sm < 4; ++sm)
#pragma unroll
        for (int sn = 0; sn < 2; ++sn)
#pragma unroll
            for (int r = 0; r < 8; ++r) {          // C layout: row = r + 8*lh
                int row = Mb + wm * 64 + sm * 16 + r + 8 * lh;
                int col = Nb + wn * 32 + sn * 16 + ln;
                if (WRITE_F32)
                    Yf[(size_t)row * D_MODEL + col] = acc[sm][sn][r];
                else
                    Yb[(size_t)row * D_MODEL + col] = to_bf16(acc[sm][sn][r]);
            }
}

// ---------------------------------------------------------------------------
// Flash attention: 1 WG = (b, h, 128 q-rows); stream 64-key blocks.
// 8 waves, each owns 16 q-rows. Scores + P*V via WMMA bf16.
// Q/K tiles arrive via TDM (DMA with row padding) when available, else async
// global->LDS; V is transposed through VGPRs.
// ---------------------------------------------------------------------------
__global__ __launch_bounds__(256)
void attention_fwd(const bf16* __restrict__ Qp, const bf16* __restrict__ Kp,
                   const bf16* __restrict__ Vp, bf16* __restrict__ Ctx) {
    constexpr int LQ = 72, LK = 72, LV = 72, LP = 72; // padded strides (16B-aligned rows)
    __shared__ bf16 Qs[128 * LQ];   // [qrow][d]
    __shared__ bf16 Ks[64 * LK];    // [key][d]
    __shared__ bf16 Vt[64 * LV];    // [d][key]  (transposed for B-layout)
    __shared__ bf16 Ps[128 * LP];   // [qrow][key]  C-layout -> A-layout relay

    const int tid  = threadIdx.x;
    const int lane = tid & 31;
    const int wave = tid >> 5;
    const int ln   = lane & 15;
    const int lh   = lane >> 4;
    const int kb8  = lh * 8;
    const int qblk = blockIdx.x;
    const int h    = blockIdx.y;
    const int bb   = blockIdx.z;
    const size_t rowQ0 = (size_t)bb * SEQ + (size_t)qblk * 128;
    const int colH = h * D_HEAD;

    // Q block load (Q already pre-scaled by 1/8 via Wq conversion)
#if USE_TDM
    if (wave == 0)
        tdm_load_2d((unsigned)(uintptr_t)&Qs[0],
                    &Qp[rowQ0 * D_MODEL + colH], D_HEAD, 128, D_MODEL);
#else
#pragma unroll
    for (int i = 0; i < 4; ++i) {
        int linear = i * 256 + tid;               // 1024 chunks of 8
        int row = linear >> 3;
        int d8  = (linear & 7) * 8;
        cp16_g2l(&Qs[row * LQ + d8],
                 &Qp[(rowQ0 + row) * D_MODEL + colH + d8]);
    }
#endif

    v8f ctx[4];
#pragma unroll
    for (int c = 0; c < 4; ++c) ctx[c] = vzero8();
    float m_loc[8], l_loc[8];
#pragma unroll
    for (int r = 0; r < 8; ++r) { m_loc[r] = -1e30f; l_loc[r] = 0.0f; }
#if USE_TDM
    if (wave == 0) tdm_wait();
#endif
    async_fence();
    __syncthreads();

    for (int kb = 0; kb < SEQ / 64; ++kb) {
        __syncthreads();  // prior block's Ks/Vt/Ps consumers done
        const size_t rowK0 = (size_t)bb * SEQ + (size_t)kb * 64;
#if USE_TDM
        if (wave == 0)
            tdm_load_2d((unsigned)(uintptr_t)&Ks[0],
                        &Kp[rowK0 * D_MODEL + colH], D_HEAD, 64, D_MODEL);
#endif
#pragma unroll
        for (int i = 0; i < 2; ++i) {             // 64x64 tile: 512 chunks of 8
            int linear = i * 256 + tid;
            int row = linear >> 3;
            int d8  = (linear & 7) * 8;
#if !USE_TDM
            cp16_g2l(&Ks[row * LK + d8],
                     &Kp[(rowK0 + row) * D_MODEL + colH + d8]);
#endif
            FragU tmp;
            tmp.q[0] = *(const u32x4*)&Vp[(rowK0 + row) * D_MODEL + colH + d8];
#pragma unroll
            for (int j = 0; j < 8; ++j) Vt[(d8 + j) * LV + row] = tmp.e[j];
        }
#if USE_TDM
        if (wave == 0) tdm_wait();
#else
        async_fence();
#endif
        __syncthreads();

        // ---- scores: wave's 16 q-rows x 64 keys = 4 WMMA tiles, K=Dh via 2 chunks
        FragU aq[2];
#pragma unroll
        for (int dc = 0; dc < 2; ++dc) {
            const bf16* base = &Qs[(wave * 16 + ln) * LQ + dc * 32];
#pragma unroll
            for (int p = 0; p < 8; ++p) {
                int K = (kb8 << 1) + (p << 1);
                aq[dc].w[p] = *(const unsigned int*)&base[K];
            }
        }
        v8f st[4];
#pragma unroll
        for (int t = 0; t < 4; ++t) {
            v8f s = vzero8();
#pragma unroll
            for (int dc = 0; dc < 2; ++dc) {
                FragU bk;
                const bf16* base = &Ks[(t * 16 + ln) * LK + dc * 32 + lh * 16];
                bk.q[0] = *(const u32x4*)&base[0];
                bk.q[1] = *(const u32x4*)&base[8];
                s = wmma_bf16(aq[dc].v, bk.v, s);
            }
            st[t] = s;
        }

        // ---- online softmax (rows live in 16-lane halves; masks < 16 only)
#pragma unroll
        for (int r = 0; r < 8; ++r) {
            float rmax = -1e30f;
#pragma unroll
            for (int t = 0; t < 4; ++t) rmax = fmaxf(rmax, st[t][r]);
#pragma unroll
            for (int m = 1; m < 16; m <<= 1)
                rmax = fmaxf(rmax, __shfl_xor(rmax, m, 32));
            float mnew  = fmaxf(m_loc[r], rmax);
            float alpha = __expf(m_loc[r] - mnew);
            float rsum  = 0.0f;
            int   prow  = wave * 16 + r + 8 * lh;
#pragma unroll
            for (int t = 0; t < 4; ++t) {
                float p = __expf(st[t][r] - mnew);
                rsum += p;
                Ps[prow * LP + t * 16 + ln] = to_bf16(p);
            }
#pragma unroll
            for (int m = 1; m < 16; m <<= 1) rsum += __shfl_xor(rsum, m, 32);
            l_loc[r] = l_loc[r] * alpha + rsum;
            m_loc[r] = mnew;
#pragma unroll
            for (int c = 0; c < 4; ++c) ctx[c][r] *= alpha;
        }
        __syncthreads();  // Ps visible

        // ---- ctx += P (16x64) x V (64x64): K chunks of 32 over keys
#pragma unroll
        for (int kc = 0; kc < 2; ++kc) {
            FragU ap;
            const bf16* base = &Ps[(wave * 16 + ln) * LP + kc * 32];
#pragma unroll
            for (int p = 0; p < 8; ++p) {
                int K = (kb8 << 1) + (p << 1);
                ap.w[p] = *(const unsigned int*)&base[K];
            }
#pragma unroll
            for (int c = 0; c < 4; ++c) {
                FragU bv;
                const bf16* vb = &Vt[(c * 16 + ln) * LV + kc * 32 + lh * 16];
                bv.q[0] = *(const u32x4*)&vb[0];
                bv.q[1] = *(const u32x4*)&vb[8];
                ctx[c] = wmma_bf16(ap.v, bv.v, ctx[c]);
            }
        }
    }

    // finalize: divide by softmax denom, write bf16 ctx in [B,S,H*Dh] layout
#pragma unroll
    for (int r = 0; r < 8; ++r) {
        float inv = 1.0f / l_loc[r];
#pragma unroll
        for (int c = 0; c < 4; ++c) {
            int row = qblk * 128 + wave * 16 + r + 8 * lh;
            int col = colH + c * 16 + ln;
            Ctx[((size_t)bb * SEQ + row) * D_MODEL + col] = to_bf16(ctx[c][r] * inv);
        }
    }
}

// ---------------------------------------------------------------------------
// out = LayerNorm(O + residual) * gamma + beta  (population var, eps=1e-6)
// one 256-thread block per row
// ---------------------------------------------------------------------------
__global__ __launch_bounds__(256)
void ln_residual(const float* __restrict__ O, const float* __restrict__ resid,
                 const float* __restrict__ gamma, const float* __restrict__ beta,
                 float* __restrict__ out) {
    const int row = blockIdx.x, tid = threadIdx.x;
    const float* o = &O[(size_t)row * D_MODEL];
    const float* q = &resid[(size_t)row * D_MODEL];
    float x[4], s = 0.0f, ss = 0.0f;
#pragma unroll
    for (int i = 0; i < 4; ++i) {
        int c = tid + i * 256;
        x[i] = o[c] + q[c];
        s += x[i];
        ss += x[i] * x[i];
    }
#pragma unroll
    for (int m = 1; m < 32; m <<= 1) {
        s  += __shfl_xor(s,  m, 32);
        ss += __shfl_xor(ss, m, 32);
    }
    __shared__ float rs[8], rss[8];
    __shared__ float mu_s, rstd_s;
    if ((tid & 31) == 0) { rs[tid >> 5] = s; rss[tid >> 5] = ss; }
    __syncthreads();
    if (tid == 0) {
        float S = 0.0f, SS = 0.0f;
#pragma unroll
        for (int i = 0; i < 8; ++i) { S += rs[i]; SS += rss[i]; }
        float mu  = S / (float)D_MODEL;
        float var = SS / (float)D_MODEL - mu * mu;
        mu_s   = mu;
        rstd_s = rsqrtf(var + 1e-6f);
    }
    __syncthreads();
    float mu = mu_s, rstd = rstd_s;
#pragma unroll
    for (int i = 0; i < 4; ++i) {
        int c = tid + i * 256;
        out[(size_t)row * D_MODEL + c] = (x[i] - mu) * rstd * gamma[c] + beta[c];
    }
}

// ---------------------------------------------------------------------------
extern "C" void kernel_launch(void* const* d_in, const int* in_sizes, int n_in,
                              void* d_out, int out_size, void* d_ws, size_t ws_size,
                              hipStream_t stream) {
    const float* q  = (const float*)d_in[0];
    const float* k  = (const float*)d_in[1];
    const float* v  = (const float*)d_in[2];
    const float* Wq = (const float*)d_in[3];
    const float* Wk = (const float*)d_in[4];
    const float* Wv = (const float*)d_in[5];
    const float* Wo = (const float*)d_in[6];
    const float* ga = (const float*)d_in[7];
    const float* be = (const float*)d_in[8];
    float* out = (float*)d_out;

    const size_t NACT = (size_t)MTOT * D_MODEL;     // 8M activations
    const size_t NW   = (size_t)D_MODEL * D_MODEL;  // 1M weights

    char*  ws  = (char*)d_ws;
    size_t off = 0;
    auto take = [&](size_t bytes) {
        void* p = ws + off;
        off += (bytes + 255) & ~(size_t)255;
        return p;
    };
    bf16* qb  = (bf16*)take(NACT * 2);
    bf16* kb  = (bf16*)take(NACT * 2);
    bf16* vb  = (bf16*)take(NACT * 2);
    bf16* wqb = (bf16*)take(NW * 2);
    bf16* wkb = (bf16*)take(NW * 2);
    bf16* wvb = (bf16*)take(NW * 2);
    bf16* wob = (bf16*)take(NW * 2);
    bf16* Qp  = (bf16*)take(NACT * 2);
    bf16* Kp  = (bf16*)take(NACT * 2);
    bf16* Vp  = (bf16*)take(NACT * 2);
    bf16* Cx  = (bf16*)take(NACT * 2);
    float* O32 = (float*)take(NACT * 4);

    // fp32 -> bf16 (fold 1/sqrt(D_HEAD)=0.125 into Wq)
    cvt_f32_bf16<<<(int)(NACT / 1024), 256, 0, stream>>>(q, qb, (int)NACT, 1.0f);
    cvt_f32_bf16<<<(int)(NACT / 1024), 256, 0, stream>>>(k, kb, (int)NACT, 1.0f);
    cvt_f32_bf16<<<(int)(NACT / 1024), 256, 0, stream>>>(v, vb, (int)NACT, 1.0f);
    cvt_f32_bf16<<<(int)(NW / 1024), 256, 0, stream>>>(Wq, wqb, (int)NW, 0.125f);
    cvt_f32_bf16<<<(int)(NW / 1024), 256, 0, stream>>>(Wk, wkb, (int)NW, 1.0f);
    cvt_f32_bf16<<<(int)(NW / 1024), 256, 0, stream>>>(Wv, wvb, (int)NW, 1.0f);
    cvt_f32_bf16<<<(int)(NW / 1024), 256, 0, stream>>>(Wo, wob, (int)NW, 1.0f);

    dim3 ggrid(MTOT / 128, D_MODEL / 128);
    gemm_xwt<0><<<ggrid, 256, 0, stream>>>(qb, wqb, Qp, nullptr);
    gemm_xwt<0><<<ggrid, 256, 0, stream>>>(kb, wkb, Kp, nullptr);
    gemm_xwt<0><<<ggrid, 256, 0, stream>>>(vb, wvb, Vp, nullptr);

    attention_fwd<<<dim3(SEQ / 128, HEADS, BATCH), 256, 0, stream>>>(Qp, Kp, Vp, Cx);

    gemm_xwt<1><<<ggrid, 256, 0, stream>>>(Cx, wob, nullptr, O32);

    ln_residual<<<MTOT, 256, 0, stream>>>(O32, q, ga, be, out);
}
